// HierarchicalArcFaceModel_50818053046757
// MI455X (gfx1250) — matrix operand order repaired and checked by reading, not compiled, not run on hardware
//
#include <hip/hip_runtime.h>
#include <hip/hip_bf16.h>

// ---------------------------------------------------------------------------
// Types
// ---------------------------------------------------------------------------
typedef __attribute__((ext_vector_type(16))) __bf16          bf16x16;
typedef __attribute__((ext_vector_type(16))) unsigned short  u16x16;
typedef __attribute__((ext_vector_type(8)))  float           v8f;
typedef __attribute__((ext_vector_type(4)))  float           f32x4;
typedef __attribute__((ext_vector_type(4)))  unsigned int    u32x4;

#define DD     512     // feature dim
#define MTILE  64      // rows per block (4 M-waves x 2 N-waves)
#define NHEAD  32      // padded head rows: 0-5 group, 8..31 = member g*4+j

// exact-truncation split: x == hi + r, lo = bf16(r). hi is exactly bf16.
__device__ __forceinline__ void split2(float x, unsigned short& hi, unsigned short& lo) {
    unsigned u = __float_as_uint(x);
    hi = (unsigned short)(u >> 16);
    float h = __uint_as_float(u & 0xffff0000u);
    float r = x - h;
    lo = (unsigned short)(__float_as_uint(r) >> 16);
}

// ---------------------------------------------------------------------------
// Preprocessing: split W_fc -> hi/lo bf16, build padded 32x512 head matrix,
// fold BatchNorm(eval) into per-column scale/bias.
// ---------------------------------------------------------------------------
__global__ __launch_bounds__(256) void prep_kernel(
    const float* __restrict__ W_fc, const float* __restrict__ b_fc,
    const float* __restrict__ gamma, const float* __restrict__ beta,
    const float* __restrict__ rmean, const float* __restrict__ rvar,
    const float* __restrict__ W_group, const float* __restrict__ W_member,
    unsigned short* __restrict__ Whi, unsigned short* __restrict__ Wlo,
    unsigned short* __restrict__ Hhi, unsigned short* __restrict__ Hlo,
    float* __restrict__ scale, float* __restrict__ bnbias)
{
    int i = blockIdx.x * blockDim.x + threadIdx.x;
    if (i < DD * DD) {
        unsigned short h, l;
        split2(W_fc[i], h, l);
        Whi[i] = h; Wlo[i] = l;
    }
    if (i < NHEAD * DD) {
        int r = i / DD, n = i - r * DD;
        float v = 0.0f;
        if (r < 6)       v = W_group[r * DD + n];
        else if (r >= 8) v = W_member[(r - 8) * DD + n];   // [6,4,512] flat
        unsigned short h, l;
        split2(v, h, l);
        Hhi[i] = h; Hlo[i] = l;
    }
    if (i < DD) {
        float s = gamma[i] * rsqrtf(rvar[i] + 1e-5f);
        scale[i]  = s;
        bnbias[i] = (b_fc[i] - rmean[i]) * s + beta[i];
    }
}

// ---------------------------------------------------------------------------
// Fragment loaders
// ---------------------------------------------------------------------------
// A fragment (16-bit A 16x32 layout): lane<16 holds row M=lane&15 with
// K = koff..koff+15 contiguous, koff = (lane<16 ? 0 : 16).
__device__ __forceinline__ void load_a_frag_f32(const float* __restrict__ row,
                                                int kb, int koff,
                                                bf16x16& Ahi, bf16x16& Alo)
{
    f32x4 a0 = *(const f32x4*)(row + kb + koff);
    f32x4 a1 = *(const f32x4*)(row + kb + koff + 4);
    f32x4 a2 = *(const f32x4*)(row + kb + koff + 8);
    f32x4 a3 = *(const f32x4*)(row + kb + koff + 12);
    u16x16 h, l;
#pragma unroll
    for (int i = 0; i < 4; ++i) {
        unsigned short hh, ll;
        split2(a0[i], hh, ll); h[i]      = hh; l[i]      = ll;
        split2(a1[i], hh, ll); h[4 + i]  = hh; l[4 + i]  = ll;
        split2(a2[i], hh, ll); h[8 + i]  = hh; l[8 + i]  = ll;
        split2(a3[i], hh, ll); h[12 + i] = hh; l[12 + i] = ll;
    }
    Ahi = __builtin_bit_cast(bf16x16, h);
    Alo = __builtin_bit_cast(bf16x16, l);
}

// A fragment from LDS features packed as uint (hi bf16 in [15:0], lo in [31:16]).
__device__ __forceinline__ void load_a_frag_packed(const unsigned int* __restrict__ rowP,
                                                   int kb, int koff,
                                                   bf16x16& Ahi, bf16x16& Alo)
{
    u32x4 p0 = *(const u32x4*)(rowP + kb + koff);
    u32x4 p1 = *(const u32x4*)(rowP + kb + koff + 4);
    u32x4 p2 = *(const u32x4*)(rowP + kb + koff + 8);
    u32x4 p3 = *(const u32x4*)(rowP + kb + koff + 12);
    u16x16 h, l;
#pragma unroll
    for (int i = 0; i < 4; ++i) {
        h[i]      = (unsigned short)(p0[i]);  l[i]      = (unsigned short)(p0[i] >> 16);
        h[4 + i]  = (unsigned short)(p1[i]);  l[4 + i]  = (unsigned short)(p1[i] >> 16);
        h[8 + i]  = (unsigned short)(p2[i]);  l[8 + i]  = (unsigned short)(p2[i] >> 16);
        h[12 + i] = (unsigned short)(p3[i]);  l[12 + i] = (unsigned short)(p3[i] >> 16);
    }
    Ahi = __builtin_bit_cast(bf16x16, h);
    Alo = __builtin_bit_cast(bf16x16, l);
}

// B fragment (16-bit B 32x16): lane n holds column N=n&15,
// K = kb + (n<16 ? 0 : 16) + 0..15 -> 16 contiguous halfs (32B).
union BU { u32x4 q[2]; u16x16 v; };
__device__ __forceinline__ bf16x16 load_b_frag(const unsigned short* __restrict__ base)
{
    BU u;
    u.q[0] = *(const u32x4*)(base);
    u.q[1] = *(const u32x4*)(base + 8);
    return __builtin_bit_cast(bf16x16, u.v);
}

#define WMMA_BF16(A, B, C) \
    __builtin_amdgcn_wmma_f32_16x16x32_bf16(false, (A), false, (B), (short)0, (C), false, false)

// ---------------------------------------------------------------------------
// Fused: GEMM(split-bf16 WMMA) -> BN+ReLU -> head GEMM(WMMA) -> argmax/gather
// Block: 64 rows x 512 cols.  Waves: 4 (M) x 2 (N), each 16x256 (16 tiles).
// ---------------------------------------------------------------------------
__global__ __launch_bounds__(256) void fused_kernel(
    const float* __restrict__ x,
    const unsigned short* __restrict__ Whi, const unsigned short* __restrict__ Wlo,
    const unsigned short* __restrict__ Hhi, const unsigned short* __restrict__ Hlo,
    const float* __restrict__ scale, const float* __restrict__ bnbias,
    const float* __restrict__ b_group, const float* __restrict__ b_member,
    float* __restrict__ out, int Btot)
{
    // 128 KB: features packed (hi bf16 | lo bf16) per element (CDNA5: 320 KB/WGP).
    // Logits (64*32 f32 = 8 KB) overlaid after all feature reads complete.
    __shared__ unsigned int featP[MTILE * DD];
    float* logits = (float*)featP;

    const int wave  = threadIdx.x >> 5;
    const int lane  = threadIdx.x & 31;
    const int lh    = lane & 15;
    const int hih   = (lane >> 4) & 1;      // lane>=16 half
    const int mwave = wave >> 1;            // 0..3 : which 16-row slice
    const int nwave = wave & 1;             // 0..1 : which 256-col slice
    const int block_row = blockIdx.x * MTILE;
    const int akoff = hih ? 16 : 0;         // A-fragment K sub-offset
    const int bkadd = hih ? 16 : 0;         // B-fragment K sub-offset

    // ---------------- main GEMM ------------------------------------------
    v8f acc[16];
#pragma unroll
    for (int t = 0; t < 16; ++t)
#pragma unroll
        for (int r = 0; r < 8; ++r) acc[t][r] = 0.0f;

    const float* xrow = x + (size_t)(block_row + mwave * 16 + lh) * DD;

    for (int ks = 0; ks < 16; ++ks) {
        const int kb = ks * 32;
        bf16x16 Ahi, Alo;
        load_a_frag_f32(xrow, kb, akoff, Ahi, Alo);
#pragma unroll
        for (int t = 0; t < 16; ++t) {
            const int n = nwave * 256 + t * 16 + lh;        // W row (output col)
            const size_t off = (size_t)n * DD + kb + bkadd; // contiguous K
            bf16x16 Bhi = load_b_frag(Whi + off);
            bf16x16 Blo = load_b_frag(Wlo + off);
            acc[t] = WMMA_BF16(Ahi, Bhi, acc[t]);   // hi*hi
            acc[t] = WMMA_BF16(Ahi, Blo, acc[t]);   // hi*lo
            acc[t] = WMMA_BF16(Alo, Bhi, acc[t]);   // lo*hi  (lo*lo dropped)
        }
    }

    // ---------------- epilogue: folded BN + ReLU -> packed LDS -------------
#pragma unroll
    for (int t = 0; t < 16; ++t) {
        const int n = nwave * 256 + t * 16 + lh;
        const float s  = scale[n];
        const float bb = bnbias[n];
#pragma unroll
        for (int r = 0; r < 8; ++r) {
            const int mm = mwave * 16 + r + (hih ? 8 : 0);  // C layout
            const float f = fmaxf(fmaf(acc[t][r], s, bb), 0.0f);
            unsigned short fh, fl;
            split2(f, fh, fl);
            featP[mm * DD + n] = (unsigned int)fh | ((unsigned int)fl << 16);
        }
    }
    __syncthreads();

    // ---------------- head GEMM: [64 x 32] = feat[64x512] @ Headsᵀ ---------
    // 8 tiles: wave -> (M slice = wave>>1, N half = wave&1); all waves busy.
    v8f hacc;
#pragma unroll
    for (int r = 0; r < 8; ++r) hacc[r] = 0.0f;
    {
        const int hm = wave >> 1;                // row slice (0..3)
        const int hn = wave & 1;                 // head half (0/1)
        const unsigned int* frow = featP + (hm * 16 + lh) * DD;
        for (int ks = 0; ks < 16; ++ks) {
            const int kb = ks * 32;
            bf16x16 Ahi, Alo;
            load_a_frag_packed(frow, kb, akoff, Ahi, Alo);
            const size_t off = (size_t)(hn * 16 + lh) * DD + kb + bkadd;
            bf16x16 Bhi = load_b_frag(Hhi + off);
            bf16x16 Blo = load_b_frag(Hlo + off);
            hacc = WMMA_BF16(Ahi, Bhi, hacc);
            hacc = WMMA_BF16(Ahi, Blo, hacc);
            hacc = WMMA_BF16(Alo, Bhi, hacc);
        }
    }
    __syncthreads();                             // all feat reads done
    {
        const int hm = wave >> 1;
        const int hn = wave & 1;
#pragma unroll
        for (int r = 0; r < 8; ++r) {
            const int mm = hm * 16 + r + (hih ? 8 : 0);
            logits[mm * NHEAD + hn * 16 + lh] = hacc[r];   // overlays featP
        }
    }
    __syncthreads();                             // logits ready

    // ---------------- bias + argmax + member gather ------------------------
    if (threadIdx.x < MTILE) {
        const int mm  = threadIdx.x;
        const int row = block_row + mm;
        float best = -3.402823466e38f;
        int bg = 0;
#pragma unroll
        for (int g = 0; g < 6; ++g) {
            float v = logits[mm * NHEAD + g] + b_group[g];
            out[(size_t)row * 6 + g] = v;
            if (v > best) { best = v; bg = g; }   // strict '>' = first max (jnp.argmax)
        }
        const size_t mbase = (size_t)Btot * 6 + (size_t)row * 4;
#pragma unroll
        for (int j = 0; j < 4; ++j) {
            out[mbase + j] = logits[mm * NHEAD + 8 + bg * 4 + j] + b_member[bg * 4 + j];
        }
    }
}

// ---------------------------------------------------------------------------
// Launch
// ---------------------------------------------------------------------------
extern "C" void kernel_launch(void* const* d_in, const int* in_sizes, int n_in,
                              void* d_out, int out_size, void* d_ws, size_t ws_size,
                              hipStream_t stream)
{
    const float* x        = (const float*)d_in[0];
    const float* W_fc     = (const float*)d_in[1];
    const float* b_fc     = (const float*)d_in[2];
    const float* gamma    = (const float*)d_in[3];
    const float* beta     = (const float*)d_in[4];
    const float* rmean    = (const float*)d_in[5];
    const float* rvar     = (const float*)d_in[6];
    const float* W_group  = (const float*)d_in[7];
    const float* b_group  = (const float*)d_in[8];
    const float* W_member = (const float*)d_in[9];
    const float* b_member = (const float*)d_in[10];
    float* out            = (float*)d_out;

    // workspace layout (~1.07 MB)
    char* ws = (char*)d_ws;
    unsigned short* Whi = (unsigned short*)(ws);
    unsigned short* Wlo = (unsigned short*)(ws + (size_t)DD * DD * 2);
    unsigned short* Hhi = (unsigned short*)(ws + (size_t)DD * DD * 4);
    unsigned short* Hlo = (unsigned short*)(ws + (size_t)DD * DD * 4 + (size_t)NHEAD * DD * 2);
    float* scale  = (float*)(ws + (size_t)DD * DD * 4 + (size_t)NHEAD * DD * 4);
    float* bnbias = scale + DD;

    const int Btot = in_sizes[0] / DD;

    prep_kernel<<<(DD * DD + 255) / 256, 256, 0, stream>>>(
        W_fc, b_fc, gamma, beta, rmean, rvar, W_group, W_member,
        Whi, Wlo, Hhi, Hlo, scale, bnbias);

    fused_kernel<<<Btot / MTILE, 256, 0, stream>>>(
        x, Whi, Wlo, Hhi, Hlo, scale, bnbias, b_group, b_member, out, Btot);
}
